// Linear_24867860644059
// MI455X (gfx1250) — matrix operand order repaired and checked by reading, not compiled, not run on hardware
//
#include <hip/hip_runtime.h>

typedef __attribute__((ext_vector_type(16))) __bf16 v16bf;
typedef __attribute__((ext_vector_type(8)))  float  v8f;
typedef __attribute__((ext_vector_type(4)))  int    i32x4;

#define BM 128
#define BN 128
#define LDA 40   // padded row pitch in bf16 elems (80B: 16B-aligned, bank-staggered)
#define LDB 40

#define ASG __attribute__((address_space(1)))   // global
#define ASL __attribute__((address_space(3)))   // LDS

#if defined(__has_builtin)
#  if __has_builtin(__builtin_amdgcn_global_load_async_to_lds_b128)
#    define HAVE_ASYNC_LDS 1
#  endif
#endif
#ifndef HAVE_ASYNC_LDS
#  define HAVE_ASYNC_LDS 0
#endif

__device__ __forceinline__ void async_wait_all() {
#if defined(__has_builtin)
#  if __has_builtin(__builtin_amdgcn_s_wait_asynccnt)
    __builtin_amdgcn_s_wait_asynccnt(0);
    return;
#  endif
#endif
    asm volatile("s_wait_asynccnt 0x0" ::: "memory");
}

// Round-to-nearest-even fp32 -> bf16 pair packed in one dword.
__device__ __forceinline__ unsigned pk_bf16(float a, float b) {
    unsigned ua = __float_as_uint(a);
    unsigned ub = __float_as_uint(b);
    ua = (ua + 0x7FFFu + ((ua >> 16) & 1u)) >> 16;
    ub = (ub + 0x7FFFu + ((ub >> 16) & 1u)) >> 16;
    return ua | (ub << 16);
}

// Dequant 8 nibbles of one qweight word to 8 bf16 via magic-exponent trick:
//   bf16(0x4300 | nib) == 128 + nib  (exact: nib fits bf16's 7+1-bit mantissa)
//   (nib - z)*s == (128+nib)*s + c,  c = -(128+z)*s
__device__ __forceinline__ uint4 dq_word(unsigned w, unsigned s2, unsigned c2) {
    unsigned p0 = ( w         & 0xFu) | ((w << 12) & 0x000F0000u) | 0x43004300u;
    unsigned p1 = ((w >>  8)  & 0xFu) | ((w <<  4) & 0x000F0000u) | 0x43004300u;
    unsigned p2 = ((w >> 16)  & 0xFu) | ((w >>  4) & 0x000F0000u) | 0x43004300u;
    unsigned p3 = ((w >> 24)  & 0xFu) | ((w >> 12) & 0x000F0000u) | 0x43004300u;
    uint4 r;
    asm("v_pk_fma_bf16 %0, %1, %2, %3" : "=v"(r.x) : "v"(p0), "v"(s2), "v"(c2));
    asm("v_pk_fma_bf16 %0, %1, %2, %3" : "=v"(r.y) : "v"(p1), "v"(s2), "v"(c2));
    asm("v_pk_fma_bf16 %0, %1, %2, %3" : "=v"(r.z) : "v"(p2), "v"(s2), "v"(c2));
    asm("v_pk_fma_bf16 %0, %1, %2, %3" : "=v"(r.w) : "v"(p3), "v"(s2), "v"(c2));
    return r;
}

union FragBF  { uint4 q[2]; v16bf v; };
union FragAcc { v8f v; float f[8]; };

// Prepass: fp32 x -> packed bf16 (8 elems per uint4).
__global__ __launch_bounds__(256)
void cvt_x_bf16(const float* __restrict__ x, uint4* __restrict__ xb) {
    const size_t i = (size_t)blockIdx.x * 256 + threadIdx.x;
    const float4* xp = (const float4*)x;
    float4 a = xp[2 * i], b = xp[2 * i + 1];
    uint4 r;
    r.x = pk_bf16(a.x, a.y); r.y = pk_bf16(a.z, a.w);
    r.z = pk_bf16(b.x, b.y); r.w = pk_bf16(b.z, b.w);
    xb[i] = r;
}

template<bool BF16A>
__global__ __launch_bounds__(256, 1)
void awq_gemm_bf16_wmma(const float* __restrict__ x,
                        const uint4* __restrict__ xb,     // used iff BF16A
                        const int*   __restrict__ qweight,
                        const float* __restrict__ scales,
                        const int*   __restrict__ qzeros,
                        const float* __restrict__ bias,
                        float* __restrict__ y,
                        int M, int K, int N)
{
    __shared__ __align__(16) unsigned short As[2][BM * LDA]; // [m][k] bf16
    __shared__ __align__(16) unsigned short Bs[2][BN * LDB]; // [n][k] bf16 (transposed)

    const int t    = threadIdx.x;
    const int lane = t & 31;
    const int wid  = t >> 5;
    const int wm   = wid >> 1;   // 4 waves along M (32 rows each)
    const int wn   = wid & 1;    // 2 waves along N (64 cols each)
    const int m0   = blockIdx.y * BM;
    const int n0   = blockIdx.x * BN;
    const int l16  = lane & 15;
    const int h    = lane >> 4;

    // W-loader role: fixed column nn; rows rr and rr+2 of each 4-word K-slab.
    const int nn   = t & 127;
    const int rr   = t >> 7;           // 0 or 1
    const int colW = n0 + nn;

    v8f acc[2][4];
    #pragma unroll
    for (int i = 0; i < 2; ++i)
        #pragma unroll
        for (int j = 0; j < 4; ++j)
            acc[i][j] = (v8f){0.f,0.f,0.f,0.f,0.f,0.f,0.f,0.f};

    const int NS = K / 32;             // 32-K slabs

    // ---- fetch-side persistent state ----
    unsigned s2 = 0, c2 = 0;           // per-group packed-bf16 scale / offset
    unsigned wq0, wq1;                 // raw qweight words
    uint4  wa[2];                      // raw A staging (only if no async path)
    float4 af[4];                      // raw A (fp32 fallback)

    // LOAD(s, buf): issue global traffic for slab s; A may go straight to LDS[buf].
    auto LOAD = [&](int s, int buf) {
        if ((s & 3) == 0) {            // AWQ group boundary (group = 4 slabs)
            const int g = s >> 2;
            float sv = scales[(size_t)g * N + colW];
            float zv = (float)qzeros[(size_t)g * N + colW];
            s2 = pk_bf16(sv, sv);
            float cv = -(zv + 128.f) * sv;
            c2 = pk_bf16(cv, cv);
        }
        wq0 = (unsigned)qweight[(size_t)(s * 4 + rr)     * N + colW];
        wq1 = (unsigned)qweight[(size_t)(s * 4 + rr + 2) * N + colW];
        const int k0 = s * 32;
        if (BF16A) {
            #pragma unroll
            for (int i = 0; i < 2; ++i) {
                const int idx = t + 256 * i;           // 512 uint4 = 128 rows x 4
                const int row = idx >> 2, kq = idx & 3;
                const uint4* gp = xb + (size_t)(m0 + row) * (K >> 3) + (k0 >> 3) + kq;
#if HAVE_ASYNC_LDS
                // CDNA5 async copy: no VGPR staging, tracked by ASYNCcnt.
                __builtin_amdgcn_global_load_async_to_lds_b128(
                    (ASG i32x4*)gp,
                    (ASL i32x4*)&As[buf][row * LDA + kq * 8], 0, 0);
#else
                wa[i] = *gp;
#endif
            }
        } else {
            #pragma unroll
            for (int i = 0; i < 4; ++i) {
                const int idx = t + 256 * i;           // 1024 float4 = 128 rows x 8
                const int row = idx >> 3, kq = idx & 7;
                af[i] = *(const float4*)(x + (size_t)(m0 + row) * K + k0 + kq * 4);
            }
        }
        (void)buf; (void)wa;
    };

    auto STORE = [&](int buf) {
        uint4 b0 = dq_word(wq0, s2, c2);
        uint4 b1 = dq_word(wq1, s2, c2);
        *(uint4*)&Bs[buf][nn * LDB +  rr      * 8] = b0;  // K = [8rr, 8rr+8)
        *(uint4*)&Bs[buf][nn * LDB + (rr + 2) * 8] = b1;  // K = [8rr+16, 8rr+24)
        if (BF16A) {
#if !HAVE_ASYNC_LDS
            #pragma unroll
            for (int i = 0; i < 2; ++i) {
                const int idx = t + 256 * i;
                const int row = idx >> 2, kq = idx & 3;
                *(uint4*)&As[buf][row * LDA + kq * 8] = wa[i];
            }
#endif
        } else {
            #pragma unroll
            for (int i = 0; i < 4; ++i) {
                const int idx = t + 256 * i;
                const int row = idx >> 3, kq = idx & 7;
                uint2 p;
                p.x = pk_bf16(af[i].x, af[i].y);
                p.y = pk_bf16(af[i].z, af[i].w);
                *(uint2*)&As[buf][row * LDA + kq * 4] = p;
            }
        }
    };

    auto FENCE = [&]() {
        if (BF16A && HAVE_ASYNC_LDS) async_wait_all();  // As writes landed in LDS
        __syncthreads();
    };

    auto COMPUTE = [&](int buf) {
        // A 16x32 bf16: lane(m=l16, half=h): VGPR0-3 = K 8h..8h+7, VGPR4-7 = K 16+8h..+7
        FragBF a[2];
        #pragma unroll
        for (int tm = 0; tm < 2; ++tm) {
            const int mrow = wm * 32 + tm * 16 + l16;
            a[tm].q[0] = *(const uint4*)&As[buf][mrow * LDA +      h * 8];
            a[tm].q[1] = *(const uint4*)&As[buf][mrow * LDA + 16 + h * 8];
        }
        // B 32x16 bf16: lane(n=l16, half=h): VGPR0-3 = K 8h..8h+7, VGPR4-7 = K 16+8h..+7
        // -> same K order as A, from the [n][k] LDS tile.
        FragBF b[4];
        #pragma unroll
        for (int tn = 0; tn < 4; ++tn) {
            const int ncol = wn * 64 + tn * 16 + l16;
            const uint4* p = (const uint4*)&Bs[buf][ncol * LDB + h * 8];
            b[tn].q[0] = p[0];
            b[tn].q[1] = p[2];
        }
        #pragma unroll
        for (int tm = 0; tm < 2; ++tm)
            #pragma unroll
            for (int tn = 0; tn < 4; ++tn)
                acc[tm][tn] = __builtin_amdgcn_wmma_f32_16x16x32_bf16(
                    false, a[tm].v, false, b[tn].v,
                    (short)0, acc[tm][tn], false, false);
    };

    // ---- software-pipelined main loop: 1 barrier per slab ----
    LOAD(0, 0);
    STORE(0);
    FENCE();
    for (int s = 0; s < NS; ++s) {
        const int cur = s & 1;
        if (s + 1 < NS) LOAD(s + 1, cur ^ 1);  // in flight over the WMMAs
        COMPUTE(cur);
        if (s + 1 < NS) {
            STORE(cur ^ 1);                    // dequant + fill the other buffer
            FENCE();
        }
    }

    // ---- epilogue: C/D layout (M = 8h + vgpr, N = l16) ----
    #pragma unroll
    for (int tm = 0; tm < 2; ++tm) {
        const int row0 = m0 + wm * 32 + tm * 16 + h * 8;
        #pragma unroll
        for (int tn = 0; tn < 4; ++tn) {
            const int col = n0 + wn * 64 + tn * 16 + l16;
            const float bb = bias[col];
            FragAcc fa; fa.v = acc[tm][tn];
            float* yp = y + (size_t)row0 * N + col;
            #pragma unroll
            for (int v = 0; v < 8; ++v)
                yp[(size_t)v * N] = fa.f[v] + bb;
        }
    }
}

extern "C" void kernel_launch(void* const* d_in, const int* in_sizes, int n_in,
                              void* d_out, int out_size, void* d_ws, size_t ws_size,
                              hipStream_t stream) {
    (void)n_in; (void)out_size;
    const float* x       = (const float*)d_in[0];
    const int*   qweight = (const int*)  d_in[1];
    const float* scales  = (const float*)d_in[2];
    const int*   qzeros  = (const int*)  d_in[3];
    const float* bias    = (const float*)d_in[4];
    float*       y       = (float*)d_out;

    // Dims from input sizes: bias=N, qweight=(K/8)*N, x=M*K
    const int N = in_sizes[4];
    const int K = (int)(8LL * in_sizes[1] / N);
    const int M = (int)((long long)in_sizes[0] / K);

    dim3 grid(N / BN, M / BM);   // x-major dispatch keeps one M-tile's x rows hot in L2

    const size_t needA = (size_t)M * K * 2;
    if (ws_size >= needA) {
        const size_t n8 = (size_t)M * K / 8;      // uint4 (8 bf16) per thread
        cvt_x_bf16<<<dim3((unsigned)(n8 / 256)), 256, 0, stream>>>(x, (uint4*)d_ws);
        awq_gemm_bf16_wmma<true><<<grid, 256, 0, stream>>>(
            x, (const uint4*)d_ws, qweight, scales, qzeros, bias, y, M, K, N);
    } else {
        awq_gemm_bf16_wmma<false><<<grid, 256, 0, stream>>>(
            x, nullptr, qweight, scales, qzeros, bias, y, M, K, N);
    }
}
